// SpatialBayesianAttention_1082331759385
// MI455X (gfx1250) — matrix operand order repaired and checked by reading, not compiled, not run on hardware
//
#include <hip/hip_runtime.h>
#include <math.h>

// ---------------------------------------------------------------------------
// SpatialBayesianAttention for MI455X (gfx1250, wave32, WMMA bf16 16x16x32)
//
// All heavy math runs through v_wmma_f32_16x16x32_bf16 with fp32 accumulation.
// Attention is flash-style (online softmax), so the 4x4096x4096 score tensor
// never touches HBM. Intermediates are bf16 in workspace, laid out so that
// WMMA A-tiles (lane=M, contiguous K chunks) are plain 16-byte vector loads;
// B-tiles (lane=N, K strided by ld) use per-element 2-byte loads.
// ---------------------------------------------------------------------------

typedef __bf16 bf16;
typedef __attribute__((ext_vector_type(16))) __bf16        v16bf;
typedef __attribute__((ext_vector_type(8)))  float         v8f;
typedef __attribute__((ext_vector_type(4)))  unsigned int  v4u;

static constexpr int BATCH = 4;
static constexpr int C_IN  = 512;
static constexpr int MID   = 128;
static constexpr int HW    = 4096;

union V16U { v4u u[2]; v16bf v; };

// A operand, 16x32 bf16 tile from row-major [M][K] (ld in elements).
// lane l: M = l&15 ; K chunks [kh..kh+7] and [16+kh..16+kh+7], kh = (l>>4)*8.
static __device__ inline v16bf load_a_16x32(const bf16* base, int ld, int lane) {
  const int m  = lane & 15;
  const int kh = (lane >> 4) << 3;
  const bf16* p = base + (size_t)m * ld + kh;
  V16U t;
  t.u[0] = *(const v4u*)(p);
  t.u[1] = *(const v4u*)(p + 16);
  return t.v;
}

// B operand, 32x16 bf16 tile from row-major [K][N] (ld in elements).
// lane l: N = l&15 ; K [kh..kh+7] and [16+kh..16+kh+7], kh = (l>>4)*8.
static __device__ inline v16bf load_b_32x16(const bf16* base, int ld, int lane) {
  const int n  = lane & 15;
  const int kh = (lane >> 4) << 3;
  v16bf t;
#pragma unroll
  for (int s = 0; s < 8; ++s) {
    t[s]     = base[(size_t)(kh + s) * ld + n];
    t[8 + s] = base[(size_t)(16 + kh + s) * ld + n];
  }
  return t;
}

static __device__ inline v8f wmma_bf16(v16bf a, v16bf b, v8f c) {
  return __builtin_amdgcn_wmma_f32_16x16x32_bf16(
      /*neg_a=*/false, a, /*neg_b=*/false, b,
      /*c_mod=*/(short)0, c, /*reuse_a=*/false, /*reuse_b=*/false);
}

// Reductions across the 16-lane half-groups (C/D rows live across 16 lanes).
static __device__ inline float red_max16(float v) {
#pragma unroll
  for (int m = 1; m < 16; m <<= 1) v = fmaxf(v, __shfl_xor(v, m, 32));
  return v;
}
static __device__ inline float red_sum16(float v) {
#pragma unroll
  for (int m = 1; m < 16; m <<= 1) v += __shfl_xor(v, m, 32);
  return v;
}

// ---------------------------------------------------------------------------
// Tiled transpose + fp32->bf16 convert: in [R][C] f32 -> out [C][R] bf16.
// grid = (C/32, R/32, slabs), block = (32, 8); slab stride = R*C.
// ---------------------------------------------------------------------------
__global__ void transpose_cvt_kernel(const float* __restrict__ in,
                                     bf16* __restrict__ out, int R, int C) {
  __shared__ float t[32][33];
  const size_t slab = (size_t)blockIdx.z * (size_t)R * (size_t)C;
  in  += slab;
  out += slab;
  const int c0 = blockIdx.x * 32, r0 = blockIdx.y * 32;
  const int tx = threadIdx.x, ty = threadIdx.y;
#pragma unroll
  for (int i = 0; i < 4; ++i)
    t[ty + 8 * i][tx] = in[(size_t)(r0 + ty + 8 * i) * C + c0 + tx];
  __syncthreads();
#pragma unroll
  for (int i = 0; i < 4; ++i)
    out[(size_t)(c0 + ty + 8 * i) * R + r0 + tx] = (bf16)t[tx][ty + 8 * i];
}

// ---------------------------------------------------------------------------
// Projection GEMM: out[n][o] (or [o][n]) = epilogue( xT[n][:] . wT[:][o] )
// xT: [B][HW][C_IN] bf16, wT: [C_IN][MID] bf16.
// mode 0: bias+BN+ReLU, store fT [B][HW][MID], *scale
// mode 1: bias+BN+ReLU, store g  [B][MID][HW]
// mode 2: bias only,    store hT [B][HW][MID]
// grid = (HW/64, B), block = 128 (4 waves, one 16-row n-tile per wave).
// ---------------------------------------------------------------------------
__global__ __launch_bounds__(128) void proj_kernel(
    const bf16* __restrict__ xT, const bf16* __restrict__ wT,
    const float* __restrict__ bias,
    const float* __restrict__ gamma, const float* __restrict__ beta,
    const float* __restrict__ mean,  const float* __restrict__ var,
    bf16* __restrict__ outp, int mode, float scale) {
  const int lane = threadIdx.x & 31;
  const int wid  = threadIdx.x >> 5;
  const int b    = blockIdx.y;
  const int n0   = (blockIdx.x * 4 + wid) * 16;
  const bf16* xA = xT + ((size_t)b * HW + n0) * C_IN;

  v8f acc[8] = {};
  for (int kk = 0; kk < C_IN; kk += 32) {
    v16bf a = load_a_16x32(xA + kk, C_IN, lane);
#pragma unroll
    for (int ct = 0; ct < 8; ++ct) {
      v16bf bm = load_b_32x16(wT + (size_t)kk * MID + ct * 16, MID, lane);
      acc[ct] = wmma_bf16(a, bm, acc[ct]);
    }
  }

  const int col = lane & 15, half = lane >> 4;
#pragma unroll
  for (int ct = 0; ct < 8; ++ct) {
    const int o = ct * 16 + col;
    const float bi = bias[o];
    float inv = 1.0f, add = 0.0f;
    if (mode != 2) {
      const float iv = gamma[o] * rsqrtf(var[o] + 1e-5f);
      inv = iv;
      add = beta[o] - mean[o] * iv;
    }
#pragma unroll
    for (int r = 0; r < 8; ++r) {
      const int n = n0 + r + 8 * half;
      float t = acc[ct][r] + bi;
      if (mode != 2) t = fmaxf(t * inv + add, 0.0f);
      t *= scale;
      if (mode == 1) outp[((size_t)b * MID + o) * HW + n] = (bf16)t;
      else           outp[((size_t)b * HW + n) * MID + o] = (bf16)t;
    }
  }
}

// ---------------------------------------------------------------------------
// Flash attention: per wave, one 16-row n-tile; loop m in 32-wide tiles.
// fT: [B][HW][MID] (scale prefolded), g: [B][MID][HW], hT: [B][HW][MID].
// zT out: [B][HW][MID] bf16 (softmax-normalized attn . h^T).
// grid = (HW/64, B), block = 128.
// ---------------------------------------------------------------------------
__global__ __launch_bounds__(128) void attn_kernel(
    const bf16* __restrict__ fT, const bf16* __restrict__ gN,
    const bf16* __restrict__ hT, bf16* __restrict__ zT) {
  __shared__ __align__(16) bf16 Pl[4][16 * 32];  // per-wave P tile (16n x 32m)
  const int lane = threadIdx.x & 31;
  const int wid  = threadIdx.x >> 5;
  const int b    = blockIdx.y;
  const int n0   = (blockIdx.x * 4 + wid) * 16;
  const bf16* fA = fT + ((size_t)b * HW + n0) * MID;
  const bf16* gb = gN + (size_t)b * MID * HW;
  const bf16* hb = hT + (size_t)b * HW * MID;
  bf16* P = Pl[wid];

  // Hoist the f^T A-tiles (16 x 128 = 4 K-steps), reused for every m-tile.
  v16bf aF[4];
#pragma unroll
  for (int k4 = 0; k4 < 4; ++k4) aF[k4] = load_a_16x32(fA + k4 * 32, MID, lane);

  v8f acc[8] = {};         // zT accumulator: 16 n x 128 c
  float mrow[8], lrow[8];  // online softmax stats, row = r + 8*(lane>>4)
#pragma unroll
  for (int r = 0; r < 8; ++r) { mrow[r] = -INFINITY; lrow[r] = 0.0f; }
  const int col = lane & 15, half = lane >> 4;

  for (int mt = 0; mt < HW; mt += 32) {
    // scores S[16n x 32m] = f^T g  (K = 128, scale folded into f)
    v8f s0 = {}, s1 = {};
#pragma unroll
    for (int k4 = 0; k4 < 4; ++k4) {
      const bf16* gk = gb + (size_t)(k4 * 32) * HW + mt;
      v16bf b0 = load_b_32x16(gk,      HW, lane);
      v16bf b1 = load_b_32x16(gk + 16, HW, lane);
      s0 = wmma_bf16(aF[k4], b0, s0);
      s1 = wmma_bf16(aF[k4], b1, s1);
    }

    __syncthreads();  // previous iteration's P reads are done
#pragma unroll
    for (int r = 0; r < 8; ++r) {
      const float tm = red_max16(fmaxf(s0[r], s1[r]));
      const float mn = fmaxf(mrow[r], tm);
      const float al = __expf(mrow[r] - mn);
      const float p0 = __expf(s0[r] - mn);
      const float p1 = __expf(s1[r] - mn);
      lrow[r] = lrow[r] * al + red_sum16(p0 + p1);
      mrow[r] = mn;
#pragma unroll
      for (int ct = 0; ct < 8; ++ct) acc[ct][r] *= al;
      const int row = r + 8 * half;            // C-layout -> LDS row-major
      P[row * 32 + col]      = (bf16)p0;
      P[row * 32 + 16 + col] = (bf16)p1;
    }
    __syncthreads();  // P tile complete

    // z^T[16n x 128c] += P[16n x 32m] . hT[m][c]
    v16bf pa = load_a_16x32(P, 32, lane);      // LDS -> A layout (ds_read_b128)
#pragma unroll
    for (int ct = 0; ct < 8; ++ct) {
      v16bf bh = load_b_32x16(hb + (size_t)mt * MID + ct * 16, MID, lane);
      acc[ct] = wmma_bf16(pa, bh, acc[ct]);
    }
  }

  // Normalize by softmax denominator and store zT as bf16.
#pragma unroll
  for (int ct = 0; ct < 8; ++ct) {
#pragma unroll
    for (int r = 0; r < 8; ++r) {
      const int n = n0 + r + 8 * half;
      const int o = ct * 16 + col;
      zT[((size_t)b * HW + n) * MID + o] = (bf16)(acc[ct][r] / lrow[r]);
    }
  }
}

// ---------------------------------------------------------------------------
// Output projection + residual: out[b][o][n] = zT[n][:].wvT[:][o] + bv[o] + x.
// zT: [B][HW][MID] bf16, wvT: [MID][C_IN] bf16.
// grid = (HW/64, C_IN/128, B), block = 128.
// ---------------------------------------------------------------------------
__global__ __launch_bounds__(128) void outproj_kernel(
    const bf16* __restrict__ zT, const bf16* __restrict__ wvT,
    const float* __restrict__ bv, const float* __restrict__ x,
    float* __restrict__ out) {
  const int lane = threadIdx.x & 31;
  const int wid  = threadIdx.x >> 5;
  const int b    = blockIdx.z;
  const int ob   = blockIdx.y * 128;
  const int n0   = (blockIdx.x * 4 + wid) * 16;
  const bf16* zA = zT + ((size_t)b * HW + n0) * MID;

  v16bf aZ[4];
#pragma unroll
  for (int k4 = 0; k4 < 4; ++k4) aZ[k4] = load_a_16x32(zA + k4 * 32, MID, lane);

  v8f acc[8] = {};
#pragma unroll
  for (int k4 = 0; k4 < 4; ++k4) {
#pragma unroll
    for (int ct = 0; ct < 8; ++ct) {
      v16bf bw = load_b_32x16(wvT + (size_t)(k4 * 32) * C_IN + ob + ct * 16,
                              C_IN, lane);
      acc[ct] = wmma_bf16(aZ[k4], bw, acc[ct]);
    }
  }

  const int col = lane & 15, half = lane >> 4;
#pragma unroll
  for (int ct = 0; ct < 8; ++ct) {
    const int o = ob + ct * 16 + col;
    const float bvv = bv[o];
#pragma unroll
    for (int r = 0; r < 8; ++r) {
      const int n = n0 + r + 8 * half;
      const size_t idx = ((size_t)b * C_IN + o) * HW + n;
      out[idx] = acc[ct][r] + bvv + x[idx];
    }
  }
}

// ---------------------------------------------------------------------------
extern "C" void kernel_launch(void* const* d_in, const int* in_sizes, int n_in,
                              void* d_out, int out_size, void* d_ws,
                              size_t ws_size, hipStream_t stream) {
  (void)in_sizes; (void)n_in; (void)out_size; (void)ws_size;
  const float* x     = (const float*)d_in[0];
  const float* wf    = (const float*)d_in[1];
  const float* bfb   = (const float*)d_in[2];
  const float* gf    = (const float*)d_in[3];
  const float* betaf = (const float*)d_in[4];
  const float* mf    = (const float*)d_in[5];
  const float* vf    = (const float*)d_in[6];
  const float* wg    = (const float*)d_in[7];
  const float* bg    = (const float*)d_in[8];
  const float* gg    = (const float*)d_in[9];
  const float* betag = (const float*)d_in[10];
  const float* mg    = (const float*)d_in[11];
  const float* vg    = (const float*)d_in[12];
  const float* wh    = (const float*)d_in[13];
  const float* bh    = (const float*)d_in[14];
  const float* wv    = (const float*)d_in[15];
  const float* bv    = (const float*)d_in[16];
  float* out = (float*)d_out;

  // Workspace layout (bf16), ~34 MB total.
  bf16* xT  = (bf16*)d_ws;                          // [B][HW][C_IN]
  bf16* fT  = xT  + (size_t)BATCH * HW * C_IN;      // [B][HW][MID]
  bf16* gN  = fT  + (size_t)BATCH * HW * MID;       // [B][MID][HW]
  bf16* hT  = gN  + (size_t)BATCH * MID * HW;       // [B][HW][MID]
  bf16* zT  = hT  + (size_t)BATCH * HW * MID;       // [B][HW][MID]
  bf16* wTf = zT  + (size_t)BATCH * HW * MID;       // [C_IN][MID]
  bf16* wTg = wTf + (size_t)C_IN * MID;
  bf16* wTh = wTg + (size_t)C_IN * MID;
  bf16* wvT = wTh + (size_t)C_IN * MID;             // [MID][C_IN]

  // 1) transposes / bf16 conversion
  transpose_cvt_kernel<<<dim3(HW / 32, C_IN / 32, BATCH), dim3(32, 8), 0,
                         stream>>>(x, xT, C_IN, HW);
  transpose_cvt_kernel<<<dim3(C_IN / 32, MID / 32, 1), dim3(32, 8), 0,
                         stream>>>(wf, wTf, MID, C_IN);
  transpose_cvt_kernel<<<dim3(C_IN / 32, MID / 32, 1), dim3(32, 8), 0,
                         stream>>>(wg, wTg, MID, C_IN);
  transpose_cvt_kernel<<<dim3(C_IN / 32, MID / 32, 1), dim3(32, 8), 0,
                         stream>>>(wh, wTh, MID, C_IN);
  transpose_cvt_kernel<<<dim3(MID / 32, C_IN / 32, 1), dim3(32, 8), 0,
                         stream>>>(wv, wvT, C_IN, MID);

  // 2) projections (softmax scale mid^-0.5 folded into f)
  const float scale = 0.08838834764831845f;
  proj_kernel<<<dim3(HW / 64, BATCH), 128, 0, stream>>>(
      xT, wTf, bfb, gf, betaf, mf, vf, fT, 0, scale);
  proj_kernel<<<dim3(HW / 64, BATCH), 128, 0, stream>>>(
      xT, wTg, bg, gg, betag, mg, vg, gN, 1, 1.0f);
  proj_kernel<<<dim3(HW / 64, BATCH), 128, 0, stream>>>(
      xT, wTh, bh, nullptr, nullptr, nullptr, nullptr, hT, 2, 1.0f);

  // 3) flash attention
  attn_kernel<<<dim3(HW / 64, BATCH), 128, 0, stream>>>(fT, gN, hT, zT);

  // 4) output projection + bias + residual
  outproj_kernel<<<dim3(HW / 64, C_IN / 128, BATCH), 128, 0, stream>>>(
      zT, wvT, bv, x, out);
}
